// DFlashAttention_75350906241580
// MI455X (gfx1250) — matrix operand order repaired and checked by reading, not compiled, not run on hardware
//
#include <hip/hip_runtime.h>
#include <hip/hip_bf16.h>

// ---------------------------------------------------------------------------
// MI455X (gfx1250) flash-attention pipeline: bf16 WMMA + Tensor Data Mover.
// Memory-bound (~0.68 GB min HBM traffic @ 23.3 TB/s ~= 29 us). Strategy:
//  - kv (268 MB) streamed ONCE via TDM: one tensor_load_to_lds per 32-key
//    chunk (2D tile 32x128 f32, stride 4096), double-buffered so the DMA
//    overlaps WMMA compute; transform (RoPE-K, V-transpose, bf16 cvt) reads
//    the staged tile from LDS.
//  - Split-KV (flash-decoding): 5 partitions x 128 (b,h) blocks = 640 blocks
//    so enough waves are in flight to saturate HBM; partials merged after.
//  - All conversions are native v_cvt_pk_bf16_f32.
// ---------------------------------------------------------------------------

typedef __bf16 bf16_t;
typedef __attribute__((ext_vector_type(16))) __bf16 v16bf;
typedef __attribute__((ext_vector_type(8)))  __bf16 v8bf;
typedef __attribute__((ext_vector_type(8)))  float  v8f;
typedef __attribute__((ext_vector_type(4)))  unsigned uint4v;
typedef __attribute__((ext_vector_type(8)))  int      int8v;
typedef __attribute__((ext_vector_type(4)))  int      int4v;

#define BATCH 4
#define QL    64
#define CTX   4096
#define TOT   4160     // CTX + QL
#define HID   4096
#define NH    32
#define HD    128
#define RMS_EPS 1e-6f
#define NSPLIT 5
#define CHUNKS_PER_SPLIT 26   // 5 * 26 * 32 == 4160
#define STG_W 132             // 128 dwords/row + 4 pad dwords (TDM pad feature)

__device__ __forceinline__ bf16_t f2bf(float f) { return (bf16_t)f; }

__device__ __forceinline__ v8f wmma_bf16(v16bf a, v16bf b, v8f c) {
  // D(16x16,f32) = A(16x32,bf16) * B(32x16,bf16) + C
  return __builtin_amdgcn_wmma_f32_16x16x32_bf16(false, a, false, b, (short)0, c,
                                                 false, false);
}

__device__ __forceinline__ v16bf combine8(v8bf lo, v8bf hi) {
  v16bf r;
#pragma unroll
  for (int i = 0; i < 8; ++i) { r[i] = lo[i]; r[i + 8] = hi[i]; }
  return r;
}

// TDM: DMA a 32-row x 128-float 2D tile (row stride 4096 floats) into LDS,
// inserting 4 pad DWORDs after every 128 DWORDs (-> 132-dword LDS rows).
// D# encoding per CDNA5 ISA ch.8 (groups 2/3 zero => 2D tensor).
// This toolchain declares the 6-arg builtin: (g0, g1, g2, g3, g4, cpol).
__device__ __forceinline__ void tdm_load_tile(const float* gsrc, unsigned lds_off) {
  unsigned long long ga = (unsigned long long)(size_t)gsrc;
  uint4v g0;
  g0[0] = 1u;                                            // count=1, user mode
  g0[1] = lds_off;                                       // lds_addr (bytes)
  g0[2] = (unsigned)(ga & 0xFFFFFFFFu);                  // global_addr[31:0]
  g0[3] = (unsigned)((ga >> 32) & 0x01FFFFFFu) | (2u << 30);  // addr[56:32], type=2
  int8v g1;
  g1[0] = (int)((2u << 16)        // data_size = 4 bytes
              | (1u << 20)        // pad_enable
              | (6u << 22)        // pad_interval: 128 DWORDs
              | (3u << 25));      // pad_amount: 4 DWORDs
  g1[1] = (int)(128u << 16);      // tensor_dim0 = 128   (bits 79:48)
  g1[2] = (int)(32u << 16);       // tensor_dim1 = 32    (bits 111:80)
  g1[3] = (int)(128u << 16);      // tile_dim0  = 128    (bits 127:112)
  g1[4] = (int)32;                // tile_dim1  = 32     (bits 143:128)
  g1[5] = (int)HID;               // tensor_dim0_stride  (bits 191:160)
  g1[6] = 0;
  g1[7] = 0;
  int4v z4 = {0, 0, 0, 0};
  int8v z8 = {0, 0, 0, 0, 0, 0, 0, 0};
  __builtin_amdgcn_tensor_load_to_lds(g0, g1, z4, z4, z8, 0);
}

// ---------------------------------------------------------------------------
// Kernel 1: q_raw = hidden_states @ Wqkv[:, 0:4096]   (M=256, K=4096, N=4096)
// ---------------------------------------------------------------------------
__global__ __launch_bounds__(128) void k_qproj(const float* __restrict__ hs,
                                               const float* __restrict__ Wqkv,
                                               float* __restrict__ qraw) {
  const int lane = threadIdx.x & 31;
  const int wave = threadIdx.x >> 5;
  const int tn = blockIdx.x * 4 + wave;   // 0..255 (N tiles)
  const int tm = blockIdx.y;              // 0..15  (M tiles)
  const int kh = lane >> 4;
  const int mr = lane & 15;
  const int col = tn * 16 + mr;
  const float* arow = hs + (size_t)(tm * 16 + mr) * HID;

  v8f acc = {};
  for (int k0 = 0; k0 < HID; k0 += 32) {
    const float4* a0 = (const float4*)(arow + k0 + 16 * kh);
    const float4* a1 = (const float4*)(arow + k0 + 16 * kh + 8);
    float4 alo0 = a0[0], alo1 = a0[1], ahi0 = a1[0], ahi1 = a1[1];
    v16bf a;
    a[0] = f2bf(alo0.x); a[1] = f2bf(alo0.y); a[2] = f2bf(alo0.z); a[3] = f2bf(alo0.w);
    a[4] = f2bf(alo1.x); a[5] = f2bf(alo1.y); a[6] = f2bf(alo1.z); a[7] = f2bf(alo1.w);
    a[8]  = f2bf(ahi0.x); a[9]  = f2bf(ahi0.y); a[10] = f2bf(ahi0.z); a[11] = f2bf(ahi0.w);
    a[12] = f2bf(ahi1.x); a[13] = f2bf(ahi1.y); a[14] = f2bf(ahi1.z); a[15] = f2bf(ahi1.w);
    v16bf b;
    const int kb = k0 + 16 * kh;
#pragma unroll
    for (int j = 0; j < 16; ++j)
      b[j] = f2bf(Wqkv[(size_t)(kb + j) * (3 * HID) + col]);
    acc = wmma_bf16(a, b, acc);
  }
#pragma unroll
  for (int r = 0; r < 8; ++r)
    qraw[(size_t)(tm * 16 + r + 8 * kh) * HID + col] = acc[r];
}

// ---------------------------------------------------------------------------
// Kernel 2: double RMS-norm + RoPE on Q, emit bf16 [b][q][h][d].
// ---------------------------------------------------------------------------
__global__ __launch_bounds__(128) void k_qnorm_rope(const float* __restrict__ qraw,
                                                    const float* __restrict__ cosb,
                                                    const float* __restrict__ sinb,
                                                    const float* __restrict__ qw,
                                                    bf16_t* __restrict__ qbf) {
  const int lane = threadIdx.x & 31;
  const int wid = blockIdx.x * (blockDim.x >> 5) + (threadIdx.x >> 5); // 0..8191
  const int h = wid & 31;
  const int q = (wid >> 5) & 63;
  const int b = wid >> 11;

  const float* src = qraw + (size_t)(b * QL + q) * HID + h * HD;
  float4 xv = *(const float4*)(src + 4 * lane);
  float4 wv = *(const float4*)(qw + 4 * lane);
  float x[4] = {xv.x, xv.y, xv.z, xv.w};
  float w[4] = {wv.x, wv.y, wv.z, wv.w};

#pragma unroll
  for (int pass = 0; pass < 2; ++pass) {
    float ss = 0.f;
#pragma unroll
    for (int j = 0; j < 4; ++j) ss += x[j] * x[j];
#pragma unroll
    for (int m = 16; m >= 1; m >>= 1) ss += __shfl_xor(ss, m, 32);
    float rs = rsqrtf(ss * (1.0f / HD) + RMS_EPS);
#pragma unroll
    for (int j = 0; j < 4; ++j) x[j] = x[j] * rs * w[j];
  }

  const int t = CTX + q;
  float4 cv = *(const float4*)(cosb + (size_t)t * HD + 4 * lane);
  float4 sv = *(const float4*)(sinb + (size_t)t * HD + 4 * lane);
  float cc[4] = {cv.x, cv.y, cv.z, cv.w};
  float sn[4] = {sv.x, sv.y, sv.z, sv.w};
  bf16_t* dst = qbf + ((size_t)((b * QL + q) * NH + h)) * HD + 4 * lane;
#pragma unroll
  for (int j = 0; j < 4; ++j) {
    float p = __shfl_xor(x[j], 16, 32);
    float rot = (lane < 16) ? -p : p;
    dst[j] = f2bf(x[j] * cc[j] + rot * sn[j]);
  }
}

// ---------------------------------------------------------------------------
// Kernel 3: split-KV flash attention with TDM double-buffered kv staging.
// Block = (b,h) x split; 4 waves x 16 Q rows.
// ---------------------------------------------------------------------------
__global__ __launch_bounds__(128) void k_attn(const float* __restrict__ th,
                                              const float* __restrict__ hs,
                                              const float* __restrict__ cosb,
                                              const float* __restrict__ sinb,
                                              const bf16_t* __restrict__ qbf,
                                              float* __restrict__ po,
                                              float* __restrict__ pm,
                                              float* __restrict__ pl) {
  __shared__ float  stage[2][32][STG_W]; // raw kv rows, TDM destination
  __shared__ bf16_t lds_k[32][136];      // roped K, row-major, padded
  __shared__ bf16_t lds_v[128][40];      // raw V, transposed [dim][key]
  __shared__ bf16_t lds_p[4][16][32];    // per-wave P re-layout scratch

  const int tid = threadIdx.x;
  const int lane = tid & 31;
  const int wave = tid >> 5;
  const int bh = blockIdx.x;             // b*32 + h
  const int split = blockIdx.y;
  const int b = bh >> 5, h = bh & 31;
  const int kh = lane >> 4;
  const int mr = lane & 15;
  const float scale = 0.08838834764831845f;   // 1/sqrt(128)

  const bf16_t* qb =
      qbf + ((size_t)((b * QL + wave * 16 + mr) * NH + h)) * HD;
  v16bf qa[4];
#pragma unroll
  for (int i = 0; i < 4; ++i) {
    const bf16_t* p0 = qb + i * 32 + 16 * kh;
    qa[i] = combine8(*(const v8bf*)p0, *(const v8bf*)(p0 + 8));
  }

  v8f o[8] = {};
  float mrow[8], lrow[8];
#pragma unroll
  for (int r = 0; r < 8; ++r) { mrow[r] = -INFINITY; lrow[r] = 0.f; }

  const int lkey = tid >> 2;          // loader: key 0..31
  const int ld0 = (tid & 3) * 32;     // loader: dim segment base
  const int pd0 = (ld0 < 64) ? ld0 + 64 : ld0 - 64;
  const float sgn = (ld0 < 64) ? -1.f : 1.f;

  const int c_begin = split * CHUNKS_PER_SPLIT;
  const int c_end = c_begin + CHUNKS_PER_SPLIT;

  // kv chunk base row: chunks 0..127 come from target_hidden, 128..129 from
  // hidden_states (no chunk straddles the boundary: 4096 = 128*32).
#define CHUNK_SRC(c)                                                       \
  (((c) * 32 < CTX) ? th + ((size_t)(b * CTX + (c) * 32)) * HID + h * HD   \
                    : hs + ((size_t)(b * QL + ((c) * 32 - CTX))) * HID + h * HD)

  if (wave == 0)
    tdm_load_tile(CHUNK_SRC(c_begin), (unsigned)(size_t)&stage[0][0][0]);

  for (int c = c_begin; c < c_end; ++c) {
    const int buf = (c - c_begin) & 1;
    if (wave == 0) __builtin_amdgcn_s_wait_tensorcnt((short)0);
    __syncthreads();                       // stage[buf] ready for all waves

    // Transform staged rows: roped K (row-major) + raw V (transposed).
    const int t = c * 32 + lkey;
    {
      const float4* s4 = (const float4*)(&stage[buf][lkey][ld0]);
      const float4* p4 = (const float4*)(&stage[buf][lkey][pd0]);
      const float4* c4 = (const float4*)(cosb + (size_t)t * HD + ld0);
      const float4* n4 = (const float4*)(sinb + (size_t)t * HD + ld0);
      if (c + 1 < c_end) {  // next chunk's cos/sin rows (global_prefetch_b8)
        __builtin_prefetch(cosb + (size_t)(t + 32) * HD + ld0, 0, 1);
        __builtin_prefetch(sinb + (size_t)(t + 32) * HD + ld0, 0, 1);
      }
      bf16_t kout[32];
#pragma unroll
      for (int j = 0; j < 8; ++j) {
        float4 xv = s4[j], pv = p4[j], cv = c4[j], sv = n4[j];
        kout[4 * j + 0] = f2bf(xv.x * cv.x + sgn * pv.x * sv.x);
        kout[4 * j + 1] = f2bf(xv.y * cv.y + sgn * pv.y * sv.y);
        kout[4 * j + 2] = f2bf(xv.z * cv.z + sgn * pv.z * sv.z);
        kout[4 * j + 3] = f2bf(xv.w * cv.w + sgn * pv.w * sv.w);
        lds_v[ld0 + 4 * j + 0][lkey] = f2bf(xv.x);
        lds_v[ld0 + 4 * j + 1][lkey] = f2bf(xv.y);
        lds_v[ld0 + 4 * j + 2][lkey] = f2bf(xv.z);
        lds_v[ld0 + 4 * j + 3][lkey] = f2bf(xv.w);
      }
#pragma unroll
      for (int i = 0; i < 4; ++i)
        *(v8bf*)&lds_k[lkey][ld0 + 8 * i] = *(const v8bf*)&kout[8 * i];
    }
    __syncthreads();                       // lds_k/lds_v ready; stage[buf] free

    if (wave == 0 && c + 1 < c_end)        // overlap next DMA with compute
      tdm_load_tile(CHUNK_SRC(c + 1), (unsigned)(size_t)&stage[buf ^ 1][0][0]);

    // Scores: S(16x32) = Q(16x128) @ K^T, as two 16-key column groups.
    v8f s0 = {}, s1 = {};
#pragma unroll
    for (int i = 0; i < 4; ++i) {
      const int db = i * 32 + 16 * kh;
      v16bf b0 = combine8(*(const v8bf*)&lds_k[mr][db],
                          *(const v8bf*)&lds_k[mr][db + 8]);
      s0 = wmma_bf16(qa[i], b0, s0);
      v16bf b1 = combine8(*(const v8bf*)&lds_k[16 + mr][db],
                          *(const v8bf*)&lds_k[16 + mr][db + 8]);
      s1 = wmma_bf16(qa[i], b1, s1);
    }

    // Online softmax (row = r + 8*kh; keys across lanes of each half-wave).
#pragma unroll
    for (int r = 0; r < 8; ++r) {
      float a0 = s0[r] * scale, a1 = s1[r] * scale;
      float rm = fmaxf(a0, a1);
#pragma unroll
      for (int m = 8; m >= 1; m >>= 1) rm = fmaxf(rm, __shfl_xor(rm, m, 32));
      float mnew = fmaxf(mrow[r], rm);
      float fs = __expf(mrow[r] - mnew);
      float p0 = __expf(a0 - mnew);
      float p1 = __expf(a1 - mnew);
      float rsum = p0 + p1;
#pragma unroll
      for (int m = 8; m >= 1; m >>= 1) rsum += __shfl_xor(rsum, m, 32);
      lrow[r] = lrow[r] * fs + rsum;
      mrow[r] = mnew;
#pragma unroll
      for (int db = 0; db < 8; ++db) o[db][r] *= fs;
      const int prow = r + 8 * kh;
      lds_p[wave][prow][mr] = f2bf(p0);
      lds_p[wave][prow][16 + mr] = f2bf(p1);
    }

    // O += P(16x32) @ V(32x128).
    v16bf pa = combine8(*(const v8bf*)&lds_p[wave][mr][16 * kh],
                        *(const v8bf*)&lds_p[wave][mr][16 * kh + 8]);
#pragma unroll
    for (int db = 0; db < 8; ++db) {
      const bf16_t* vp = &lds_v[db * 16 + mr][16 * kh];
      v16bf bv = combine8(*(const v8bf*)vp, *(const v8bf*)(vp + 8));
      o[db] = wmma_bf16(pa, bv, o[db]);
    }
    // next iteration's wait + barrier orders lds_k/v WAR and stage reuse
  }
#undef CHUNK_SRC

  // Partial epilogue: unnormalized O (f32) + per-row stats.
  float* op = po + (size_t)(bh * NSPLIT + split) * QL * HD;
#pragma unroll
  for (int db = 0; db < 8; ++db)
#pragma unroll
    for (int r = 0; r < 8; ++r)
      op[(size_t)(wave * 16 + r + 8 * kh) * HD + db * 16 + mr] = o[db][r];
  if (mr == 0) {
    float* mp = pm + (size_t)(bh * NSPLIT + split) * QL;
    float* lp = pl + (size_t)(bh * NSPLIT + split) * QL;
#pragma unroll
    for (int r = 0; r < 8; ++r) {
      mp[wave * 16 + r + 8 * kh] = mrow[r];
      lp[wave * 16 + r + 8 * kh] = lrow[r];
    }
  }
}

// ---------------------------------------------------------------------------
// Kernel 3b: merge the NSPLIT partials per (b,h,q) row, emit bf16 attn out.
// ---------------------------------------------------------------------------
__global__ __launch_bounds__(128) void k_combine(const float* __restrict__ po,
                                                 const float* __restrict__ pm,
                                                 const float* __restrict__ pl,
                                                 bf16_t* __restrict__ attn) {
  const int lane = threadIdx.x & 31;
  const int wid = blockIdx.x * (blockDim.x >> 5) + (threadIdx.x >> 5); // 0..8191
  const int q = wid & 63;
  const int bh = wid >> 6;      // b*32 + h

  float mi[NSPLIT], li[NSPLIT];
  float mstar = -INFINITY;
#pragma unroll
  for (int s = 0; s < NSPLIT; ++s) {
    mi[s] = pm[(size_t)(bh * NSPLIT + s) * QL + q];
    li[s] = pl[(size_t)(bh * NSPLIT + s) * QL + q];
    mstar = fmaxf(mstar, mi[s]);
  }
  float L = 0.f, ws[NSPLIT];
#pragma unroll
  for (int s = 0; s < NSPLIT; ++s) {
    ws[s] = __expf(mi[s] - mstar);
    L += ws[s] * li[s];
  }
  float acc[4] = {0.f, 0.f, 0.f, 0.f};
#pragma unroll
  for (int s = 0; s < NSPLIT; ++s) {
    const float4 ov = *(const float4*)(po +
        ((size_t)(bh * NSPLIT + s) * QL + q) * HD + 4 * lane);
    acc[0] += ws[s] * ov.x; acc[1] += ws[s] * ov.y;
    acc[2] += ws[s] * ov.z; acc[3] += ws[s] * ov.w;
  }
  const int b = bh >> 5, h = bh & 31;
  const float inv = 1.0f / L;
  bf16_t* dst = attn + (size_t)(b * QL + q) * HID + h * HD + 4 * lane;
#pragma unroll
  for (int j = 0; j < 4; ++j) dst[j] = f2bf(acc[j] * inv);
}

// ---------------------------------------------------------------------------
// Kernel 4: out = attn(256x4096, bf16) @ Wo(4096x4096, f32) -> f32
// ---------------------------------------------------------------------------
__global__ __launch_bounds__(128) void k_oproj(const bf16_t* __restrict__ attn,
                                               const float* __restrict__ Wo,
                                               float* __restrict__ out) {
  const int lane = threadIdx.x & 31;
  const int wave = threadIdx.x >> 5;
  const int tn = blockIdx.x * 4 + wave;
  const int tm = blockIdx.y;
  const int kh = lane >> 4;
  const int mr = lane & 15;
  const int col = tn * 16 + mr;
  const bf16_t* arow = attn + (size_t)(tm * 16 + mr) * HID;

  v8f acc = {};
  for (int k0 = 0; k0 < HID; k0 += 32) {
    v16bf a = combine8(*(const v8bf*)(arow + k0 + 16 * kh),
                       *(const v8bf*)(arow + k0 + 16 * kh + 8));
    v16bf b;
    const int kb = k0 + 16 * kh;
#pragma unroll
    for (int j = 0; j < 16; ++j)
      b[j] = f2bf(Wo[(size_t)(kb + j) * HID + col]);
    acc = wmma_bf16(a, b, acc);
  }
#pragma unroll
  for (int r = 0; r < 8; ++r)
    out[(size_t)(tm * 16 + r + 8 * kh) * HID + col] = acc[r];
}

// ---------------------------------------------------------------------------
extern "C" void kernel_launch(void* const* d_in, const int* in_sizes, int n_in,
                              void* d_out, int out_size, void* d_ws, size_t ws_size,
                              hipStream_t stream) {
  const float* hs   = (const float*)d_in[0];   // hidden_states  (4,64,4096)
  const float* th   = (const float*)d_in[1];   // target_hidden  (4,4096,4096)
  const float* cosb = (const float*)d_in[2];   // (4160,128)
  const float* sinb = (const float*)d_in[3];   // (4160,128)
  const float* Wqkv = (const float*)d_in[4];   // (4096,12288)
  const float* Wo   = (const float*)d_in[5];   // (4096,4096)
  const float* qw   = (const float*)d_in[6];   // (128,)
  float* out = (float*)d_out;                  // (4,64,4096) f32

  char* ws = (char*)d_ws;
  const size_t OFF_QRAW = 0;                                  // 4 MB
  const size_t OFF_QBF  = (size_t)4 << 20;                    // 2 MB
  const size_t OFF_ATTN = (size_t)6 << 20;                    // 2 MB
  const size_t OFF_PO   = (size_t)8 << 20;                    // 20 MB partial O
  const size_t PO_BYTES = (size_t)BATCH * NH * NSPLIT * QL * HD * 4;
  const size_t OFF_PM   = OFF_PO + PO_BYTES;
  const size_t PM_BYTES = (size_t)BATCH * NH * NSPLIT * QL * 4;
  const size_t OFF_PL   = OFF_PM + PM_BYTES;

  float*  qraw = (float*)(ws + OFF_QRAW);
  bf16_t* qbf  = (bf16_t*)(ws + OFF_QBF);
  bf16_t* attn = (bf16_t*)(ws + OFF_ATTN);
  float*  po   = (float*)(ws + OFF_PO);
  float*  pm   = (float*)(ws + OFF_PM);
  float*  pl   = (float*)(ws + OFF_PL);

  k_qproj     <<<dim3(64, 16),             128, 0, stream>>>(hs, Wqkv, qraw);
  k_qnorm_rope<<<dim3(2048),               128, 0, stream>>>(qraw, cosb, sinb, qw, qbf);
  k_attn      <<<dim3(BATCH * NH, NSPLIT), 128, 0, stream>>>(th, hs, cosb, sinb,
                                                             qbf, po, pm, pl);
  k_combine   <<<dim3(2048),               128, 0, stream>>>(po, pm, pl, attn);
  k_oproj     <<<dim3(64, 16),             128, 0, stream>>>(attn, Wo, out);
}
